// JointAttentionMemoryBank_13984413515814
// MI455X (gfx1250) — matrix operand order repaired and checked by reading, not compiled, not run on hardware
//
#include <hip/hip_runtime.h>
#include <hip/hip_bf16.h>

// ---------- types ----------
typedef __attribute__((ext_vector_type(16))) __bf16          v16bf;
typedef __attribute__((ext_vector_type(8)))  float           v8f;
typedef __attribute__((ext_vector_type(8)))  unsigned short  ushort8;
typedef int v4i_ __attribute__((vector_size(16)));           // async builtin's pointee

union BV {            // bf16 WMMA operand viewed as two 16-byte halves
    v16bf   bf;
    ushort8 h[2];
    unsigned short us[16];
};

__device__ __forceinline__ unsigned short f2bf(float f) {
    unsigned u = __builtin_bit_cast(unsigned, f);
    unsigned r = u + 0x7FFFu + ((u >> 16) & 1u);   // round-to-nearest-even
    return (unsigned short)(r >> 16);
}

static constexpr int Bb = 16, Nn = 4096, Dd = 128, Mm = 1536;
static constexpr int MC = 64;                        // M-chunk per iteration
static constexpr int NCHUNK = Mm / MC;               // 24
static constexpr float SCALE = 0.08838834764831845f; // 1/sqrt(128)

// ---------- LDS layout (shorts) ----------
static constexpr int LRA  = Dd + 8;    // 136: wT row pitch (272B)
static constexpr int LRB  = MC + 8;    // 72 : wB row pitch (144B)
static constexpr int LROW = MC + 8;    // 72 : P row pitch (wave-private)
static constexpr int WT_SZ = MC * LRA;          // 8704
static constexpr int WB_SZ = Dd * LRB;          // 9216
static constexpr int OFF_WT = 0;                // two buffers
static constexpr int OFF_WB = 2 * WT_SZ;        // two buffers
static constexpr int OFF_P  = OFF_WB + 2 * WB_SZ;
static constexpr int LDS_TOTAL = OFF_P + 4 * 16 * LROW;

// ---------- async copy (gfx1250) with safe fallback ----------
#if defined(__has_builtin)
#if __has_builtin(__builtin_amdgcn_global_load_async_to_lds_b128)
#define HAVE_ASYNC_LDS 1
#endif
#endif

#ifdef HAVE_ASYNC_LDS
#define ASYNC_B128(gp, lp)                                                     \
    __builtin_amdgcn_global_load_async_to_lds_b128(                            \
        (__attribute__((address_space(1))) v4i_*)(gp),                         \
        (__attribute__((address_space(3))) v4i_*)(lp), 0, 0)
#if __has_builtin(__builtin_amdgcn_s_wait_asynccnt)
#define WAIT_ASYNC() __builtin_amdgcn_s_wait_asynccnt(0)
#else
#define WAIT_ASYNC() asm volatile("s_wait_asynccnt 0x0" ::: "memory")
#endif
#else
#define ASYNC_B128(gp, lp) (*(ushort8*)(lp) = *(const ushort8*)(gp))
#define WAIT_ASYNC() ((void)0)
#endif

// ---------- prep: f32 w -> bf16 wT (MxD) and wB (DxM) ----------
__global__ void prep_w(const float* __restrict__ mem,
                       unsigned short* __restrict__ wT,
                       unsigned short* __restrict__ wB) {
    int i = blockIdx.x * 256 + threadIdx.x;
    if (i >= Dd * Mm) return;
    int d = i / Mm, m = i - d * Mm;
    unsigned short v = f2bf(mem[i]);   // mem is (D, M) row-major
    wB[i] = v;                         // (D, M)
    wT[m * Dd + d] = v;                // (M, D)
}

// ---------- fragment loaders (from LDS) ----------
// A-operand (16 rows x 32 K): lane r rows; lanes 0-15 K 0..15,
// lanes 16-31 K 16..31.
__device__ __forceinline__ v16bf load_tileA_lds(const unsigned short* rowbase,
                                                int pitch, int r, int hi) {
    const ushort8* p = (const ushort8*)(rowbase + r * pitch + hi * 16);
    BV u; u.h[0] = p[0]; u.h[1] = p[1];            // +0 and +8 elements
    return u.bf;
}

// B-operand (32 K x 16 cols): lane n K 0..15, lane n+16 K 16..31, K contiguous.
__device__ __forceinline__ v16bf load_tileB_lds(const unsigned short* colbase,
                                                int hi) {
    const ushort8* p = (const ushort8*)(colbase + hi * 16);
    BV u; u.h[0] = p[0]; u.h[1] = p[1];
    return u.bf;
}

#define WMMA_BF16(A, B, C) \
    __builtin_amdgcn_wmma_f32_16x16x32_bf16(false, (A), false, (B), (short)0, (C), false, false)

// cooperative staging of one M-chunk of w into LDS (coalesced b128 per lane)
__device__ __forceinline__ void prefetch_chunk(const unsigned short* __restrict__ wT,
                                               const unsigned short* __restrict__ wB,
                                               int mc, unsigned short* lds, int buf,
                                               int tid) {
    unsigned short* wTb = lds + OFF_WT + buf * WT_SZ;
    unsigned short* wBb = lds + OFF_WB + buf * WB_SZ;
#pragma unroll
    for (int i = 0; i < 8; ++i) {                 // wT chunk: 64 rows x 128 sh
        int e = tid + i * 128;
        int row = e >> 4, c = e & 15;
        ASYNC_B128(wT + (size_t)(mc + row) * Dd + c * 8, wTb + row * LRA + c * 8);
    }
#pragma unroll
    for (int i = 0; i < 8; ++i) {                 // wB chunk: 128 rows x 64 sh
        int e = tid + i * 128;
        int row = e >> 3, c = e & 7;
        ASYNC_B128(wB + (size_t)row * Mm + mc + c * 8, wBb + row * LRB + c * 8);
    }
}

__global__ void __launch_bounds__(128, 1)
fused_membank_attn(const float* __restrict__ x,
                   const unsigned short* __restrict__ wT,
                   const unsigned short* __restrict__ wB,
                   float* __restrict__ out) {
    __shared__ __align__(16) unsigned short lds[LDS_TOTAL];

    const int tid  = threadIdx.x;
    const int wv   = tid >> 5;
    const int lane = tid & 31;
    const int r    = lane & 15;          // row (A) / column (B,C) index
    const int hi   = lane >> 4;
    const int b    = blockIdx.y;
    const int n0   = blockIdx.x * 64 + wv * 16;
    const int ncol = n0 + r;             // this lane's output column

    unsigned short* myP = lds + OFF_P + wv * 16 * LROW;

    // ---- load & convert this wave's x tile (16 cols x 128 d) into B-regs ----
    v16bf xr[4];
#pragma unroll
    for (int kc = 0; kc < 4; ++kc) {
        const float* xs = x + ((size_t)b * Nn + ncol) * Dd + kc * 32 + hi * 16;
        BV u;
#pragma unroll
        for (int j = 0; j < 16; ++j) u.us[j] = f2bf(xs[j]);
        xr[kc] = u.bf;
    }

    // ---- online-softmax state ----
    float mx = -1e30f, sm = 0.0f;
    v8f O[8];
#pragma unroll
    for (int dt = 0; dt < 8; ++dt) O[dt] = (v8f)0.0f;

    prefetch_chunk(wT, wB, 0, lds, 0, tid);       // prologue

    for (int ci = 0; ci < NCHUNK; ++ci) {
        const int buf = ci & 1;
        WAIT_ASYNC();                 // my share of chunk ci landed
        __syncthreads();              // everyone's share landed; prev buf free
        if (ci + 1 < NCHUNK)
            prefetch_chunk(wT, wB, (ci + 1) * MC, lds, buf ^ 1, tid);

        const unsigned short* wTb = lds + OFF_WT + buf * WT_SZ;
        const unsigned short* wBb = lds + OFF_WB + buf * WB_SZ;

        // ===== GEMM1: S chunk (64 x 16), 1-ahead pipelined A loads =====
        v8f st[4];
        v16bf aN = load_tileA_lds(wTb, LRA, r, hi);
#pragma unroll
        for (int t = 0; t < 16; ++t) {
            const int mt = t >> 2, kc = t & 3;
            v16bf aC = aN;
            if (t + 1 < 16) {
                const int mt1 = (t + 1) >> 2, kc1 = (t + 1) & 3;
                aN = load_tileA_lds(wTb + mt1 * 16 * LRA + kc1 * 32, LRA, r, hi);
            }
            st[mt] = WMMA_BF16(aC, xr[kc], (kc == 0) ? (v8f)0.0f : st[mt]);
        }

        // ===== online softmax over this chunk (scale folded into exp) =====
        float cm = -1e30f;
#pragma unroll
        for (int mt = 0; mt < 4; ++mt)
#pragma unroll
            for (int j = 0; j < 8; ++j) cm = fmaxf(cm, st[mt][j]);
        cm = fmaxf(cm, __shfl_xor(cm, 16, 32));
        float nm    = fmaxf(mx, cm * SCALE);
        float alpha = __expf(mx - nm);
        mx = nm;

        float ps = 0.0f;
#pragma unroll
        for (int mt = 0; mt < 4; ++mt)
#pragma unroll
            for (int j = 0; j < 8; ++j) {
                float p = __expf(fmaf(st[mt][j], SCALE, -nm));
                st[mt][j] = p;
                ps += p;
            }

        // stage P (bf16) into wave-private LDS first, hide stores under VALU
#pragma unroll
        for (int mt = 0; mt < 4; ++mt) {
            uint4 pk;
            unsigned q[4];
#pragma unroll
            for (int j = 0; j < 4; ++j)
                q[j] = (unsigned)f2bf(st[mt][2 * j]) |
                       ((unsigned)f2bf(st[mt][2 * j + 1]) << 16);
            pk.x = q[0]; pk.y = q[1]; pk.z = q[2]; pk.w = q[3];
            *(uint4*)(myP + r * LROW + mt * 16 + hi * 8) = pk;
        }

        ps += __shfl_xor(ps, 16, 32);
        sm = sm * alpha + ps;
#pragma unroll
        for (int dt = 0; dt < 8; ++dt)
#pragma unroll
            for (int j = 0; j < 8; ++j) O[dt][j] *= alpha;

        asm volatile("s_wait_dscnt 0x0" ::: "memory");   // cross-lane handoff

        // ===== GEMM2: O (128 x 16) += wB(chunk) x P, pipelined =====
        v16bf pb0 = load_tileB_lds(myP + r * LROW, hi);
        v16bf pb1 = load_tileB_lds(myP + r * LROW + 32, hi);
        v16bf a2N = load_tileA_lds(wBb, LRB, r, hi);
#pragma unroll
        for (int t = 0; t < 16; ++t) {
            const int kc2 = t >> 3, dt = t & 7;
            v16bf aC = a2N;
            if (t + 1 < 16) {
                const int kc2n = (t + 1) >> 3, dtn = (t + 1) & 7;
                a2N = load_tileA_lds(wBb + dtn * 16 * LRB + kc2n * 32, LRB, r, hi);
            }
            O[dt] = WMMA_BF16(aC, (kc2 == 0) ? pb0 : pb1, O[dt]);
        }
    }

    // ---- normalize and write out (B, N, D) ----
    float inv = 1.0f / sm;
    float* orow = out + ((size_t)b * Nn + ncol) * Dd;
#pragma unroll
    for (int dt = 0; dt < 8; ++dt)
#pragma unroll
        for (int j = 0; j < 8; ++j)
            orow[dt * 16 + hi * 8 + j] = O[dt][j] * inv;
}

extern "C" void kernel_launch(void* const* d_in, const int* in_sizes, int n_in,
                              void* d_out, int out_size, void* d_ws, size_t ws_size,
                              hipStream_t stream) {
    const float* x   = (const float*)d_in[0];   // (B, N, D) f32
    const float* mem = (const float*)d_in[1];   // (1, D, M) f32
    float* out = (float*)d_out;                 // (B, N, D) f32

    unsigned short* wT = (unsigned short*)d_ws;            // (M, D) bf16
    unsigned short* wB = wT + (size_t)Mm * Dd;             // (D, M) bf16

    int total = Dd * Mm;
    prep_w<<<(total + 255) / 256, 256, 0, stream>>>(mem, wT, wB);

    dim3 grid(Nn / 64, Bb);
    fused_membank_attn<<<grid, 128, 0, stream>>>(x, wT, wB, out);
}
